// BitBLASW8A8Linear_34127810134610
// MI455X (gfx1250) — 8-bit matrix operand order repaired and checked by reading, compiled, not run on hardware
//
#include <hip/hip_runtime.h>

typedef __attribute__((ext_vector_type(8))) int v8i;
typedef __attribute__((ext_vector_type(4))) int v4i;

#define TOKENS 4096
#define NFEAT  4096
#define KFEAT  4096
#define BM 128
#define BN 128
#define BK 64
#define BUFBYTES (BM * BK)

#if __has_builtin(__builtin_amdgcn_global_load_async_to_lds_b128) && \
    __has_builtin(__builtin_amdgcn_s_wait_asynccnt)
#define HAVE_ASYNC_LDS 1
typedef __attribute__((address_space(1))) v4i gv4i;
typedef __attribute__((address_space(3))) v4i lv4i;
#else
#define HAVE_ASYNC_LDS 0
#endif

__global__ __launch_bounds__(256) void absmax_kernel(const float* __restrict__ x,
                                                     unsigned* __restrict__ amax,
                                                     int n4) {
    __shared__ float red[256];
    float m = 0.0f;
    const float4* x4 = (const float4*)x;
    for (int i = blockIdx.x * blockDim.x + threadIdx.x; i < n4;
         i += gridDim.x * blockDim.x) {
        float4 v = x4[i];
        m = fmaxf(m, fmaxf(fmaxf(fabsf(v.x), fabsf(v.y)),
                           fmaxf(fabsf(v.z), fabsf(v.w))));
    }
    red[threadIdx.x] = m;
    __syncthreads();
    for (int s = 128; s > 0; s >>= 1) {
        if ((int)threadIdx.x < s)
            red[threadIdx.x] = fmaxf(red[threadIdx.x], red[threadIdx.x + s]);
        __syncthreads();
    }
    if (threadIdx.x == 0) atomicMax(amax, __float_as_uint(red[0]));
}

__global__ __launch_bounds__(256) void quant_kernel(const float* __restrict__ x,
                                                    const unsigned* __restrict__ amax,
                                                    signed char* __restrict__ xq,
                                                    int n4) {
    float absmax = __uint_as_float(*amax);
    float sinv = 127.0f / (absmax + 1e-8f);
    int i = blockIdx.x * blockDim.x + threadIdx.x;
    if (i < n4) {
        float4 v = ((const float4*)x)[i];
        char4 q;
        q.x = (signed char)(int)fminf(fmaxf(rintf(v.x * sinv), -127.0f), 127.0f);
        q.y = (signed char)(int)fminf(fmaxf(rintf(v.y * sinv), -127.0f), 127.0f);
        q.z = (signed char)(int)fminf(fmaxf(rintf(v.z * sinv), -127.0f), 127.0f);
        q.w = (signed char)(int)fminf(fmaxf(rintf(v.w * sinv), -127.0f), 127.0f);
        ((char4*)xq)[i] = q;
    }
}

__global__ __launch_bounds__(256) void gemm_kernel(const signed char* __restrict__ xq,
                                                   const signed char* __restrict__ w,
                                                   const float* __restrict__ wscale,
                                                   const unsigned* __restrict__ amax,
                                                   float* __restrict__ out) {
    __shared__ __align__(16) signed char lA[2 * BUFBYTES];
    __shared__ __align__(16) signed char lB[2 * BUFBYTES];

    const int tid  = threadIdx.x;
    const int lane = tid & 31;
    const int wid  = tid >> 5;
    const int wm   = wid >> 1;
    const int wn   = wid & 1;
    const int mi   = lane & 15;
    const int hi   = lane >> 4;

    const int m0 = blockIdx.y * BM;
    const int n0 = blockIdx.x * BN;

    const int lrow  = tid >> 1;
    const int loff  = (tid & 1) * 32;
    const int stoff = lrow * BK + loff;

    v8i acc[2][4];
    const v8i vzero = {0, 0, 0, 0, 0, 0, 0, 0};
#pragma unroll
    for (int mt = 0; mt < 2; ++mt)
#pragma unroll
        for (int nt = 0; nt < 4; ++nt) acc[mt][nt] = vzero;

    const signed char* gA = xq + (size_t)(m0 + lrow) * KFEAT + loff;
    const signed char* gB = w  + (size_t)(n0 + lrow) * KFEAT + loff;

    const int ksteps = KFEAT / BK;

    const int afr = (wm * 32 + mi) * BK;
    const int bfr = (wn * 64 + mi) * BK;

#if HAVE_ASYNC_LDS
    __builtin_amdgcn_global_load_async_to_lds_b128((gv4i*)(gA),
                                                   (lv4i*)(&lA[stoff]), 0, 0);
    __builtin_amdgcn_global_load_async_to_lds_b128((gv4i*)(gA + 16),
                                                   (lv4i*)(&lA[stoff + 16]), 0, 0);
    __builtin_amdgcn_global_load_async_to_lds_b128((gv4i*)(gB),
                                                   (lv4i*)(&lB[stoff]), 0, 0);
    __builtin_amdgcn_global_load_async_to_lds_b128((gv4i*)(gB + 16),
                                                   (lv4i*)(&lB[stoff + 16]), 0, 0);
    __builtin_amdgcn_s_wait_asynccnt(0);
    __syncthreads();
#else
    {
        uint4 ra0 = *(const uint4*)(gA);
        uint4 ra1 = *(const uint4*)(gA + 16);
        uint4 rb0 = *(const uint4*)(gB);
        uint4 rb1 = *(const uint4*)(gB + 16);
        *(uint4*)(&lA[stoff])      = ra0;
        *(uint4*)(&lA[stoff + 16]) = ra1;
        *(uint4*)(&lB[stoff])      = rb0;
        *(uint4*)(&lB[stoff + 16]) = rb1;
        __syncthreads();
    }
#endif

    const signed char* pA = gA + BK;
    const signed char* pB = gB + BK;
    int nxtoff = BUFBYTES;

#pragma clang loop unroll(disable)
    for (int kt = 0; kt < ksteps; ++kt) {
        const int curoff = nxtoff ^ BUFBYTES;
        const bool more  = (kt + 1) < ksteps;

#if HAVE_ASYNC_LDS
        if (more) {
            __builtin_amdgcn_global_load_async_to_lds_b128(
                (gv4i*)(pA), (lv4i*)(&lA[nxtoff + stoff]), 0, 0);
            __builtin_amdgcn_global_load_async_to_lds_b128(
                (gv4i*)(pA + 16), (lv4i*)(&lA[nxtoff + stoff + 16]), 0, 0);
            __builtin_amdgcn_global_load_async_to_lds_b128(
                (gv4i*)(pB), (lv4i*)(&lB[nxtoff + stoff]), 0, 0);
            __builtin_amdgcn_global_load_async_to_lds_b128(
                (gv4i*)(pB + 16), (lv4i*)(&lB[nxtoff + stoff + 16]), 0, 0);
        }
#endif

        v8i afrag[2];
#pragma unroll
        for (int mt = 0; mt < 2; ++mt) {
            const uint2* ar =
                (const uint2*)(&lA[curoff + afr + mt * (16 * BK)]);
            uint2 p0 = ar[2 * hi + 0];
            uint2 p1 = ar[2 * hi + 1];
            uint2 p2 = ar[2 * hi + 4];
            uint2 p3 = ar[2 * hi + 5];
            v8i a = {(int)p0.x, (int)p0.y, (int)p1.x, (int)p1.y,
                     (int)p2.x, (int)p2.y, (int)p3.x, (int)p3.y};
            afrag[mt] = a;
        }

        v8i bfrag[4];
#pragma unroll
        for (int nt = 0; nt < 4; ++nt) {
            const uint4* br =
                (const uint4*)(&lB[curoff + bfr + nt * (16 * BK)]);
            uint4 q0 = br[hi];
            uint4 q1 = br[2 + hi];
            v8i b = {(int)q0.x, (int)q0.y, (int)q0.z, (int)q0.w,
                     (int)q1.x, (int)q1.y, (int)q1.z, (int)q1.w};
            bfrag[nt] = b;
        }

#pragma unroll
        for (int mt = 0; mt < 2; ++mt)
#pragma unroll
            for (int nt = 0; nt < 4; ++nt)
                acc[mt][nt] = __builtin_amdgcn_wmma_i32_16x16x64_iu8(
                    true, afrag[mt], true, bfrag[nt], acc[mt][nt],
                    false, false);

#if HAVE_ASYNC_LDS
        if (more) __builtin_amdgcn_s_wait_asynccnt(0);
        __syncthreads();
#else
        if (more) {
            uint4 ra0 = *(const uint4*)(pA);
            uint4 ra1 = *(const uint4*)(pA + 16);
            uint4 rb0 = *(const uint4*)(pB);
            uint4 rb1 = *(const uint4*)(pB + 16);
            __syncthreads();
            *(uint4*)(&lA[nxtoff + stoff])      = ra0;
            *(uint4*)(&lA[nxtoff + stoff + 16]) = ra1;
            *(uint4*)(&lB[nxtoff + stoff])      = rb0;
            *(uint4*)(&lB[nxtoff + stoff + 16]) = rb1;
        }
        __syncthreads();
#endif

        pA += BK;
        pB += BK;
        nxtoff ^= BUFBYTES;
    }

    const float act_scale = __uint_as_float(*amax) / 127.0f;
#pragma unroll
    for (int nt = 0; nt < 4; ++nt) {
        const int col = n0 + wn * 64 + nt * 16 + mi;
        const float s = act_scale * wscale[col];
#pragma unroll
        for (int mt = 0; mt < 2; ++mt) {
            const int rowbase = m0 + wm * 32 + mt * 16 + hi * 8;
            v8i c = acc[mt][nt];
#pragma unroll
            for (int v = 0; v < 8; ++v)
                out[(size_t)(rowbase + v) * NFEAT + col] = (float)c[v] * s;
        }
    }
}

extern "C" void kernel_launch(void* const* d_in, const int* in_sizes, int n_in,
                              void* d_out, int out_size, void* d_ws, size_t ws_size,
                              hipStream_t stream) {
    const float*       x      = (const float*)d_in[0];
    const signed char* qw     = (const signed char*)d_in[1];
    const float*       wscale = (const float*)d_in[2];
    float*             out    = (float*)d_out;

    unsigned*    amax = (unsigned*)d_ws;
    signed char* xq   = (signed char*)d_ws + 256;

    (void)in_sizes; (void)n_in; (void)out_size; (void)ws_size;

    const int n4 = (TOKENS * KFEAT) / 4;

    (void)hipMemsetAsync(d_ws, 0, 4, stream);
    absmax_kernel<<<1024, 256, 0, stream>>>(x, amax, n4);
    quant_kernel<<<n4 / 256, 256, 0, stream>>>(x, amax, xq, n4);

    dim3 grid(NFEAT / BN, TOKENS / BM);
    gemm_kernel<<<grid, 256, 0, stream>>>(xq, qw, wscale, amax, out);
}
